// ResidualGNNs_24163486007511
// MI455X (gfx1250) — matrix operand order repaired and checked by reading, not compiled, not run on hardware
//
#include <hip/hip_runtime.h>
#include <math.h>

// ---------- problem constants ----------
#define NODES   400
#define GNUM    128
#define NTOT    (GNUM * NODES)      // 51200
#define DEG     16
#define EEDGE   (NTOT * DEG)        // 819200
#define NE      (EEDGE + NTOT)      // edges + self loops = 870400
#define HC      64
#define NL      3
#define HID     128
#define IN_DIM  79800
#define IN1     79992
#define ZKP     80000               // IN1 padded to /32
#define KP1     416                 // 400 padded to /32
#define EPS     1e-5f
#define SLOPE   0.2f

typedef __bf16 bf16_t;
typedef __attribute__((ext_vector_type(16))) __bf16 v16bf;
typedef __attribute__((ext_vector_type(8)))  __bf16 v8bf;
typedef __attribute__((ext_vector_type(8)))  float  v8f;

// ---------- helpers ----------
__device__ __forceinline__ unsigned enc_f32(float f) {
  unsigned u = __float_as_uint(f);
  return (u & 0x80000000u) ? ~u : (u | 0x80000000u);
}
__device__ __forceinline__ float dec_f32(unsigned u) {
  unsigned v = (u & 0x80000000u) ? (u ^ 0x80000000u) : ~u;
  return __uint_as_float(v);
}

// ---------- bf16 operand staging ----------
// A: row-major [rows, K] fp32 -> row-major [rows, Kp] bf16 (zero pad K..Kp)
__global__ void k_conv_a(const float* __restrict__ A, bf16_t* __restrict__ out,
                         int rows, int K, int Kp) {
  size_t i = (size_t)blockIdx.x * blockDim.x + threadIdx.x;
  size_t tot = (size_t)rows * (size_t)Kp;
  if (i >= tot) return;
  int k = (int)(i % (size_t)Kp);
  size_t r = i / (size_t)Kp;
  out[i] = (k < K) ? (bf16_t)A[r * (size_t)K + k] : (bf16_t)0.0f;
}

// B: row-major [K, N] fp32 -> transposed row-major [N, Kp] bf16 (zero pad)
__global__ void k_conv_bt(const float* __restrict__ B, bf16_t* __restrict__ out,
                          int K, int N, int Kp) {
  size_t i = (size_t)blockIdx.x * blockDim.x + threadIdx.x;
  size_t tot = (size_t)N * (size_t)Kp;
  if (i >= tot) return;
  int k = (int)(i % (size_t)Kp);
  int n = (int)(i / (size_t)Kp);
  out[i] = (k < K) ? (bf16_t)B[(size_t)k * N + n] : (bf16_t)0.0f;
}

// ---------- WMMA GEMM: C[M,N] = A[M,Kp]bf16 @ B^T[N,Kp]bf16 (+bias) ----------
// one wave per 16x16 output tile, K-loop in steps of 32 via v_wmma_f32_16x16x32_bf16
__global__ void __launch_bounds__(32)
k_gemm_bf16(const bf16_t* __restrict__ A, const bf16_t* __restrict__ Bt,
            const float* __restrict__ bias, float* __restrict__ C,
            int M, int N, int Kp) {
  const int tilesN = N >> 4;
  const int id = blockIdx.x;
  const int tm = id / tilesN;
  const int tn = id % tilesN;
  const int lane = threadIdx.x & 31;
  const int l15  = lane & 15;
  const int half = lane >> 4;              // 0 or 1
  const int mrow = (tm << 4) + l15;        // A row for this lane
  const int ncol = (tn << 4) + l15;        // B column for this lane
  const bf16_t* arow = A  + (size_t)mrow * Kp + (half << 3);   // +0 or +8
  const bf16_t* brow = Bt + (size_t)ncol * Kp + (half << 3);   // +0 or +8
  v8f acc = {};
  for (int k0 = 0; k0 < Kp; k0 += 32) {
    v8bf alo = *(const v8bf*)(arow + k0);        // K: khalf + 0..7
    v8bf ahi = *(const v8bf*)(arow + k0 + 16);   // K: khalf + 16..23
    v16bf a = __builtin_shufflevector(alo, ahi, 0,1,2,3,4,5,6,7,8,9,10,11,12,13,14,15);
    v8bf blo = *(const v8bf*)(brow + k0);        // K: khalf + 0..7
    v8bf bhi = *(const v8bf*)(brow + k0 + 16);   // K: khalf + 16..23
    v16bf b = __builtin_shufflevector(blo, bhi, 0,1,2,3,4,5,6,7,8,9,10,11,12,13,14,15);
    acc = __builtin_amdgcn_wmma_f32_16x16x32_bf16(
        false, a, false, b, (short)0, acc, false, false);
  }
  const float bv = bias ? bias[(tn << 4) + l15] : 0.0f;
  const int mbase = (tm << 4) + (half << 3);
  const int nc = (tn << 4) + l15;
#pragma unroll
  for (int i = 0; i < 8; ++i)
    C[(size_t)(mbase + i) * N + nc] = acc[i] + bv;
}

// ---------- GAT attention pipeline ----------
__global__ void k_attn_logits(const float* __restrict__ xh,
                              const float* __restrict__ as_,
                              const float* __restrict__ ad_,
                              float* __restrict__ als, float* __restrict__ ald,
                              int n) {
  int i = blockIdx.x * blockDim.x + threadIdx.x;
  if (i >= n) return;
  const float* row = xh + (size_t)i * HC;
  float s = 0.f, d = 0.f;
#pragma unroll 8
  for (int c = 0; c < HC; ++c) { float v = row[c]; s += v * as_[c]; d += v * ad_[c]; }
  als[i] = s; ald[i] = d;
}

__global__ void k_seg_init(unsigned* __restrict__ amax, float* __restrict__ den,
                           float* __restrict__ agg, int n) {
  size_t i = (size_t)blockIdx.x * blockDim.x + threadIdx.x;
  if (i < (size_t)n * HC) agg[i] = 0.0f;
  if (i < (size_t)n) { amax[i] = enc_f32(-__builtin_inff()); den[i] = 0.0f; }
}

__global__ void k_edge_max(const int* __restrict__ src, const int* __restrict__ dst,
                           const float* __restrict__ als, const float* __restrict__ ald,
                           float* __restrict__ alpha, unsigned* __restrict__ amax,
                           int ne) {
  int e = blockIdx.x * blockDim.x + threadIdx.x;
  if (e >= ne) return;
  int s, d;
  if (e < EEDGE) { s = src[e]; d = dst[e]; } else { s = d = e - EEDGE; }
  float a = als[s] + ald[d];
  a = (a > 0.f) ? a : a * SLOPE;               // leaky_relu
  alpha[e] = a;
  atomicMax(&amax[d], enc_f32(a));
}

__global__ void k_edge_exp(const int* __restrict__ dst,
                           float* __restrict__ alpha,
                           const unsigned* __restrict__ amax,
                           float* __restrict__ den, int ne) {
  int e = blockIdx.x * blockDim.x + threadIdx.x;
  if (e >= ne) return;
  int d = (e < EEDGE) ? dst[e] : (e - EEDGE);
  float ex = expf(alpha[e] - dec_f32(amax[d]));
  alpha[e] = ex;
  atomicAdd(&den[d], ex);
}

// one wave per edge, 2 channels per lane
__global__ void __launch_bounds__(256)
k_edge_msg(const int* __restrict__ src, const int* __restrict__ dst,
           const float* __restrict__ alpha, const float* __restrict__ den,
           const float* __restrict__ xh, float* __restrict__ agg, int ne) {
  int e = blockIdx.x * 8 + (threadIdx.x >> 5);
  if (e >= ne) return;
  int lane = threadIdx.x & 31;
  int s, d;
  if (e < EEDGE) { s = src[e]; d = dst[e]; } else { s = d = e - EEDGE; }
  float w = alpha[e] / den[d];
  int c = lane * 2;
  atomicAdd(&agg[(size_t)d * HC + c],     w * xh[(size_t)s * HC + c]);
  atomicAdd(&agg[(size_t)d * HC + c + 1], w * xh[(size_t)s * HC + c + 1]);
}

__global__ void k_post_tanh(const float* __restrict__ agg,
                            const float* __restrict__ bias,
                            float* __restrict__ out, size_t tot) {
  size_t i = (size_t)blockIdx.x * blockDim.x + threadIdx.x;
  if (i >= tot) return;
  int c = (int)(i % HC);
  out[i] = tanhf(agg[i] + bias[c]);
}

__global__ void __launch_bounds__(64)
k_pool(const float* __restrict__ xn, float* __restrict__ pooled) {
  int g = blockIdx.x, c = threadIdx.x;
  const float* base = xn + (size_t)g * NODES * HC + c;
  float s = 0.f;
  for (int i = 0; i < NODES; ++i) s += base[(size_t)i * HC];
  pooled[g * HC + c] = s * (1.0f / NODES);
}

// ---------- feature head ----------
// strict upper triangle extraction + BatchNorm over 128 graphs, fused into bf16 z
__global__ void k_x0_bn(const float* __restrict__ x, const float* __restrict__ gam,
                        const float* __restrict__ bet, bf16_t* __restrict__ zbf) {
  int i = blockIdx.x * blockDim.x + threadIdx.x;
  if (i >= NODES * NODES) return;
  int r = i / NODES, c = i % NODES;
  if (c <= r) return;
  int k = r * (NODES - 1) - (r * (r - 1)) / 2 + (c - r - 1);
  float m = 0.f;
  for (int g = 0; g < GNUM; ++g) m += x[((size_t)(g * NODES + r)) * NODES + c];
  m *= (1.0f / GNUM);
  float v = 0.f;
  for (int g = 0; g < GNUM; ++g) {
    float t = x[((size_t)(g * NODES + r)) * NODES + c] - m; v += t * t;
  }
  v *= (1.0f / GNUM);
  float rs = rsqrtf(v + EPS);
  float gk = gam[k], bk = bet[k];
  for (int g = 0; g < GNUM; ++g) {
    float t = x[((size_t)(g * NODES + r)) * NODES + c];
    zbf[(size_t)g * ZKP + k] = (bf16_t)((t - m) * rs * gk + bk);
  }
}

// BN over pooled conv features -> z columns IN_DIM..IN1; zero the Kp pad
__global__ void k_h_bn(const float* __restrict__ pooled, const float* __restrict__ gam,
                       const float* __restrict__ bet, bf16_t* __restrict__ zbf) {
  int j = blockIdx.x * blockDim.x + threadIdx.x;
  if (j >= (ZKP - IN_DIM)) return;       // 200 columns incl. 8 pad
  if (j >= NL * HC) {                    // pad columns 79992..79999
    for (int g = 0; g < GNUM; ++g) zbf[(size_t)g * ZKP + IN_DIM + j] = (bf16_t)0.0f;
    return;
  }
  int l = j / HC, c = j % HC;
  const float* col = pooled + (size_t)l * GNUM * HC + c;
  float m = 0.f;
  for (int g = 0; g < GNUM; ++g) m += col[(size_t)g * HC];
  m *= (1.0f / GNUM);
  float v = 0.f;
  for (int g = 0; g < GNUM; ++g) { float t = col[(size_t)g * HC] - m; v += t * t; }
  v *= (1.0f / GNUM);
  float rs = rsqrtf(v + EPS);
  float gk = gam[j], bk = bet[j];
  for (int g = 0; g < GNUM; ++g)
    zbf[(size_t)g * ZKP + IN_DIM + j] = (bf16_t)((col[(size_t)g * HC] - m) * rs * gk + bk);
}

// BatchNorm (over 128 rows) + ReLU; optional bf16 copy for next GEMM. In-place safe.
__global__ void k_bn_relu(float* __restrict__ z, const float* __restrict__ gam,
                          const float* __restrict__ bet, bf16_t* __restrict__ zb,
                          int C) {
  int c = blockIdx.x * blockDim.x + threadIdx.x;
  if (c >= C) return;
  float m = 0.f;
  for (int g = 0; g < GNUM; ++g) m += z[(size_t)g * C + c];
  m *= (1.0f / GNUM);
  float v = 0.f;
  for (int g = 0; g < GNUM; ++g) { float t = z[(size_t)g * C + c] - m; v += t * t; }
  v *= (1.0f / GNUM);
  float rs = rsqrtf(v + EPS);
  float gk = gam[c], bk = bet[c];
  for (int g = 0; g < GNUM; ++g) {
    float y = (z[(size_t)g * C + c] - m) * rs * gk + bk;
    y = (y > 0.f) ? y : 0.f;
    z[(size_t)g * C + c] = y;
    if (zb) zb[(size_t)g * C + c] = (bf16_t)y;
  }
}

__global__ void k_final(const float* __restrict__ z3, const float* __restrict__ w4,
                        const float* __restrict__ b4, float* __restrict__ out) {
  int t = blockIdx.x * blockDim.x + threadIdx.x;
  if (t >= GNUM * 2) return;
  int g = t >> 1, j = t & 1;
  float s = b4[j];
#pragma unroll 8
  for (int c = 0; c < 64; ++c) s += z3[(size_t)g * 64 + c] * w4[c * 2 + j];
  out[t] = s;
}

// ---------- host side ----------
static inline unsigned gsz(size_t n, int b) { return (unsigned)((n + (size_t)b - 1) / (size_t)b); }

extern "C" void kernel_launch(void* const* d_in, const int* in_sizes, int n_in,
                              void* d_out, int out_size, void* d_ws, size_t ws_size,
                              hipStream_t stream) {
  (void)in_sizes; (void)n_in; (void)out_size; (void)ws_size;
  const float* x     = (const float*)d_in[0];
  const int*   ei    = (const int*)d_in[1];
  const int*   src   = ei;
  const int*   dst   = ei + EEDGE;
  const float* W[3]  = {(const float*)d_in[3], (const float*)d_in[4], (const float*)d_in[5]};
  const float* att_s = (const float*)d_in[6];
  const float* att_d = (const float*)d_in[7];
  const float* cbias = (const float*)d_in[8];
  const float* bn_g  = (const float*)d_in[9];
  const float* bn_b  = (const float*)d_in[10];
  const float* bnh_g = (const float*)d_in[11];
  const float* bnh_b = (const float*)d_in[12];
  const float* mw1   = (const float*)d_in[13];
  const float* mb1   = (const float*)d_in[14];
  const float* g1    = (const float*)d_in[15];
  const float* b1    = (const float*)d_in[16];
  const float* mw2   = (const float*)d_in[17];
  const float* mb2   = (const float*)d_in[18];
  const float* g2    = (const float*)d_in[19];
  const float* b2    = (const float*)d_in[20];
  const float* mw3   = (const float*)d_in[21];
  const float* mb3   = (const float*)d_in[22];
  const float* g3    = (const float*)d_in[23];
  const float* b3    = (const float*)d_in[24];
  const float* mw4   = (const float*)d_in[25];
  const float* mb4   = (const float*)d_in[26];

  // workspace carve (256B aligned)
  char* p = (char*)d_ws;
  auto carve = [&](size_t bytes) -> void* {
    void* r = (void*)p; p += (bytes + 255) & ~(size_t)255; return r;
  };
  bf16_t*   xbf    = (bf16_t*)carve((size_t)NTOT * KP1 * 2);
  bf16_t*   wtbf   = (bf16_t*)carve((size_t)64 * KP1 * 2);
  bf16_t*   w1bf   = (bf16_t*)carve((size_t)HID * ZKP * 2);
  bf16_t*   zbf    = (bf16_t*)carve((size_t)GNUM * ZKP * 2);
  float*    xh     = (float*)carve((size_t)NTOT * HC * 4);
  float*    hbuf   = (float*)carve((size_t)NTOT * HC * 4);
  float*    agg    = (float*)carve((size_t)NTOT * HC * 4);
  float*    als    = (float*)carve((size_t)NTOT * 4);
  float*    ald    = (float*)carve((size_t)NTOT * 4);
  unsigned* amax   = (unsigned*)carve((size_t)NTOT * 4);
  float*    den    = (float*)carve((size_t)NTOT * 4);
  float*    alpha  = (float*)carve((size_t)NE * 4);
  float*    pooled = (float*)carve((size_t)NL * GNUM * HC * 4);
  float*    z1     = (float*)carve((size_t)GNUM * HID * 4);
  bf16_t*   z1bf   = (bf16_t*)carve((size_t)GNUM * HID * 2);
  float*    z2     = (float*)carve((size_t)GNUM * 64 * 4);
  bf16_t*   z2bf   = (bf16_t*)carve((size_t)GNUM * 64 * 2);
  float*    z3     = (float*)carve((size_t)GNUM * 64 * 4);

  // ---- GAT layers ----
  for (int l = 0; l < NL; ++l) {
    const int K  = (l == 0) ? NODES : HC;
    const int Kp = (l == 0) ? KP1   : HC;
    const float* Ain = (l == 0) ? x : hbuf;

    k_conv_a <<<gsz((size_t)NTOT * Kp, 256), 256, 0, stream>>>(Ain, xbf, NTOT, K, Kp);
    k_conv_bt<<<gsz((size_t)HC * Kp,   256), 256, 0, stream>>>(W[l], wtbf, K, HC, Kp);
    k_gemm_bf16<<<(NTOT / 16) * (HC / 16), 32, 0, stream>>>(xbf, wtbf, nullptr, xh, NTOT, HC, Kp);

    k_attn_logits<<<gsz(NTOT, 256), 256, 0, stream>>>(xh, att_s + l * HC, att_d + l * HC,
                                                      als, ald, NTOT);
    k_seg_init<<<gsz((size_t)NTOT * HC, 256), 256, 0, stream>>>(amax, den, agg, NTOT);
    k_edge_max<<<gsz(NE, 256), 256, 0, stream>>>(src, dst, als, ald, alpha, amax, NE);
    k_edge_exp<<<gsz(NE, 256), 256, 0, stream>>>(dst, alpha, amax, den, NE);
    k_edge_msg<<<gsz(NE, 8), 256, 0, stream>>>(src, dst, alpha, den, xh, agg, NE);
    k_post_tanh<<<gsz((size_t)NTOT * HC, 256), 256, 0, stream>>>(agg, cbias + l * HC,
                                                                 hbuf, (size_t)NTOT * HC);
    k_pool<<<GNUM, HC, 0, stream>>>(hbuf, pooled + (size_t)l * GNUM * HC);
  }

  // ---- head: build z (bf16, K padded to 80000) ----
  k_x0_bn<<<gsz(NODES * NODES, 256), 256, 0, stream>>>(x, bn_g, bn_b, zbf);
  k_h_bn<<<1, 256, 0, stream>>>(pooled, bnh_g, bnh_b, zbf);

  // ---- MLP ----
  k_conv_bt<<<gsz((size_t)HID * ZKP, 256), 256, 0, stream>>>(mw1, w1bf, IN1, HID, ZKP);
  k_gemm_bf16<<<(GNUM / 16) * (HID / 16), 32, 0, stream>>>(zbf, w1bf, mb1, z1, GNUM, HID, ZKP);
  k_bn_relu<<<1, HID, 0, stream>>>(z1, g1, b1, z1bf, HID);

  k_conv_bt<<<gsz((size_t)64 * HID, 256), 256, 0, stream>>>(mw2, wtbf, HID, 64, HID);
  k_gemm_bf16<<<(GNUM / 16) * (64 / 16), 32, 0, stream>>>(z1bf, wtbf, mb2, z2, GNUM, 64, HID);
  k_bn_relu<<<1, 64, 0, stream>>>(z2, g2, b2, z2bf, 64);

  k_conv_bt<<<gsz((size_t)64 * 64, 256), 256, 0, stream>>>(mw3, wtbf, 64, 64, 64);
  k_gemm_bf16<<<(GNUM / 16) * (64 / 16), 32, 0, stream>>>(z2bf, wtbf, mb3, z3, GNUM, 64, 64);
  k_bn_relu<<<1, 64, 0, stream>>>(z3, g3, b3, (bf16_t*)nullptr, 64);

  k_final<<<1, 256, 0, stream>>>(z3, mw4, mb4, (float*)d_out);
}
